// SineMLP_54941221651078
// MI455X (gfx1250) — matrix operand order repaired and checked by reading, not compiled, not run on hardware
//
#include <hip/hip_runtime.h>
#include <hip/hip_bf16.h>

typedef __attribute__((ext_vector_type(16))) _Float16 v16h;
typedef __attribute__((ext_vector_type(8)))  _Float16 v8h;
typedef __attribute__((ext_vector_type(8)))  float    v8f;

#define HID   256
#define ROWS  256          // rows per block: 8 waves x 32 rows
#define MT    2            // M-tiles (16 rows each) per wave
#define WH    128          // weight rows per half-layer staging buffer
#define LDSW  264          // padded row stride (halves): bank-conflict free
#define OMEGA0 30.0f

__device__ __forceinline__ void async_copy_b128(unsigned lds_off, const void* gptr) {
    unsigned long long ga = (unsigned long long)gptr;
    asm volatile("global_load_async_to_lds_b128 %0, %1, off"
                 :: "v"(lds_off), "v"(ga) : "memory");
}

__device__ __forceinline__ void wait_async_all() {
#if __has_builtin(__builtin_amdgcn_s_wait_asynccnt)
    __builtin_amdgcn_s_wait_asynccnt(0);
#else
    asm volatile("s_wait_asynccnt 0" ::: "memory");
#endif
}

__device__ __forceinline__ v8f wmma_f16(v16h a, v16h b, v8f c) {
    return __builtin_amdgcn_wmma_f32_16x16x32_f16(false, a, false, b, (short)0, c,
                                                  false, false);
}

// Pre-pass: convert W1..W3 (f32, 3 x 256 x 256) to f16 once into d_ws.
__global__ void convert_w_kernel(const float* __restrict__ W1,
                                 const float* __restrict__ W2,
                                 const float* __restrict__ W3,
                                 _Float16* __restrict__ dst) {
    const int idx = blockIdx.x * 256 + threadIdx.x;        // 0 .. 196607
    const float* srcs[3] = {W1, W2, W3};
    dst[idx] = (_Float16)srcs[idx >> 16][idx & 65535];
}

template <bool F16W>
__global__ __launch_bounds__(256, 1)
void siren_mlp_kernel(const float* __restrict__ x,
                      const float* __restrict__ W0, const float* __restrict__ b0,
                      const float* __restrict__ W1, const float* __restrict__ b1,
                      const float* __restrict__ W2, const float* __restrict__ b2,
                      const float* __restrict__ W3, const float* __restrict__ b3,
                      const float* __restrict__ W4, const float* __restrict__ b4,
                      const _Float16* __restrict__ Wf16,
                      float* __restrict__ out, int npoints)
{
    __shared__ _Float16 sW[2][WH * LDSW];       // double-buffered half-layer weights (2x66KB)
    __shared__ _Float16 sH[ROWS * LDSW];        // activations, wave-private 32-row slices (132KB)
    __shared__ float    sX[ROWS * 3];
    __shared__ float    sW4[3 * HID];
    __shared__ float    sBias[3 * HID];

    const int tid    = threadIdx.x;
    const int lane   = tid & 31;
    const int waveId = tid >> 5;
    const int laneLo = lane & 15;
    const int hiHalf = lane >> 4;
    const int blockRow0 = blockIdx.x * ROWS;

    // ---- async-DMA the input tile: 256 rows x 3 f32 = 3KB = 192 x b128 ----
    if (tid < 192) {
        unsigned lX = (unsigned)(unsigned long long)(void*)&sX[tid * 4];
        async_copy_b128(lX, x + (size_t)blockRow0 * 3 + tid * 4);
    }

    // ---- stage small constants ----
    for (int i = tid; i < 3 * HID; i += 256) sW4[i] = W4[i];
    sBias[tid]           = b1[tid];
    sBias[HID + tid]     = b2[tid];
    sBias[2 * HID + tid] = b3[tid];
    const float w0a = W0[tid * 3 + 0], w0b = W0[tid * 3 + 1], w0c = W0[tid * 3 + 2];
    const float bb0 = b0[tid];

    // ---- weight half-layer staging (async DMA when f16 weights available) ----
    const float* Wsrc3[3] = {W1, W2, W3};
    auto stage = [&](int sN, int dstBuf) {
        const int layerN = sN >> 1, halfN = sN & 1;
        if (F16W) {
            const _Float16* src = Wf16 + layerN * (HID * HID) + halfN * (WH * HID);
            unsigned ldsBase = (unsigned)(unsigned long long)(void*)&sW[dstBuf][0];
#pragma unroll
            for (int i = 0; i < 8; ++i) {                 // 2048 x 16B chunks / 256 thr
                int c = tid + i * 256;
                int row = c >> 5, col8 = (c & 31) * 8;
                async_copy_b128(ldsBase + (unsigned)((row * LDSW + col8) * 2),
                                src + row * HID + col8);
            }
        } else {
            const float* src = Wsrc3[layerN] + halfN * (WH * HID);
            for (int i = tid * 4; i < WH * HID; i += 1024) {
                float4 wv = *(const float4*)(src + i);
                int row = i >> 8, colc = i & 255;
                _Float16* dst = &sW[dstBuf][row * LDSW + colc];
                dst[0] = (_Float16)wv.x; dst[1] = (_Float16)wv.y;
                dst[2] = (_Float16)wv.z; dst[3] = (_Float16)wv.w;
            }
        }
    };

    stage(0, 0);                 // layer1/half0 DMA, drained by the wait below
    wait_async_all();            // x tile ready (weight DMA also counted; acceptable)
    __syncthreads();

    // ---- Layer 0 (K=3, VALU): thread = output unit, loop rows ----
    for (int r = 0; r < ROWS; ++r) {
        float z = sX[r * 3 + 0] * w0a + sX[r * 3 + 1] * w0b + sX[r * 3 + 2] * w0c + bb0;
        sH[r * LDSW + tid] = (_Float16)__sinf(OMEGA0 * z);
    }

    // ---- Hidden layers: 6 pipeline stages = 3 layers x 2 half-layers ----
    v16h afrag[MT][8];
    int buf = 0;
    for (int s = 0; s < 6; ++s) {
        const int layer = s >> 1, half = s & 1;

        wait_async_all();        // current buf staged
        __syncthreads();         // all waves done with buf^1 (consumed at s-1)
        if (s < 5) stage(s + 1, buf ^ 1);   // DMA next half-layer during compute

        if (half == 0) {
            // Load A fragments for this layer: wave-private rows, K=256.
            const int aoff = hiHalf ? 16 : 0;
#pragma unroll
            for (int m2 = 0; m2 < MT; ++m2) {
                const _Float16* rowp = &sH[(waveId * 32 + m2 * 16 + laneLo) * LDSW];
#pragma unroll
                for (int kt = 0; kt < 8; ++kt) {
                    v8h lo = *(const v8h*)(rowp + kt * 32 + aoff);
                    v8h hh = *(const v8h*)(rowp + kt * 32 + 8 + aoff);
                    afrag[m2][kt] = __builtin_shufflevector(lo, hh,
                        0, 1, 2, 3, 4, 5, 6, 7, 8, 9, 10, 11, 12, 13, 14, 15);
                }
            }
        }

        const int colBase = half * WH;
#pragma unroll 2
        for (int ntl = 0; ntl < 8; ++ntl) {
            const int nb = ntl * 16;
            v8f acc0 = {};          // inline SRC2=0 on first WMMA, bias in epilogue
            v8f acc1 = {};
            const _Float16* wrow = &sW[buf][(nb + laneLo) * LDSW + (hiHalf ? 16 : 0)];
#pragma unroll
            for (int kt = 0; kt < 8; ++kt) {
                v16h bfrag = *(const v16h*)(wrow + kt * 32);
                acc0 = wmma_f16(afrag[0][kt], bfrag, acc0);
                acc1 = wmma_f16(afrag[1][kt], bfrag, acc1);
            }
            const int col  = colBase + nb + laneLo;
            const float bias = sBias[layer * HID + col];
#pragma unroll
            for (int r = 0; r < 8; ++r) {
                const int srow0 = waveId * 32 + (hiHalf ? 8 : 0);
                sH[(srow0 + r) * LDSW + col]      = (_Float16)__sinf(acc0[r] + bias);
                sH[(srow0 + 16 + r) * LDSW + col] = (_Float16)__sinf(acc1[r] + bias);
            }
        }
        buf ^= 1;
    }
    __syncthreads();

    // ---- Final layer (N=3, VALU): thread = row ----
    {
        const _Float16* hrow = &sH[tid * LDSW];
        float a0 = b4[0], a1 = b4[1], a2 = b4[2];
        for (int i = 0; i < HID; ++i) {
            float hv = (float)hrow[i];
            a0 += hv * sW4[0 * HID + i];
            a1 += hv * sW4[1 * HID + i];
            a2 += hv * sW4[2 * HID + i];
        }
        const int gr = blockRow0 + tid;
        out[gr * 3 + 0] = a0;
        out[gr * 3 + 1] = a1;
        out[gr * 3 + 2] = a2;
    }
}

extern "C" void kernel_launch(void* const* d_in, const int* in_sizes, int n_in,
                              void* d_out, int out_size, void* d_ws, size_t ws_size,
                              hipStream_t stream) {
    const float* x  = (const float*)d_in[0];
    const float* W0 = (const float*)d_in[1];
    const float* b0 = (const float*)d_in[2];
    const float* W1 = (const float*)d_in[3];
    const float* b1 = (const float*)d_in[4];
    const float* W2 = (const float*)d_in[5];
    const float* b2 = (const float*)d_in[6];
    const float* W3 = (const float*)d_in[7];
    const float* b3 = (const float*)d_in[8];
    const float* W4 = (const float*)d_in[9];
    const float* b4 = (const float*)d_in[10];
    float* out = (float*)d_out;

    const int npoints = in_sizes[0] / 3;
    const int nblocks = npoints / ROWS;
    const size_t f16_bytes = (size_t)3 * HID * HID * sizeof(_Float16);

    if (ws_size >= f16_bytes) {
        _Float16* wf16 = (_Float16*)d_ws;
        hipLaunchKernelGGL(convert_w_kernel, dim3(3 * HID * HID / 256), dim3(256), 0,
                           stream, W1, W2, W3, wf16);
        hipLaunchKernelGGL((siren_mlp_kernel<true>), dim3(nblocks), dim3(256), 0, stream,
                           x, W0, b0, W1, b1, W2, b2, W3, b3, W4, b4, wf16, out, npoints);
    } else {
        hipLaunchKernelGGL((siren_mlp_kernel<false>), dim3(nblocks), dim3(256), 0, stream,
                           x, W0, b0, W1, b1, W2, b2, W3, b3, W4, b4,
                           (const _Float16*)nullptr, out, npoints);
    }
}
